// CausalExpert_67723044324028
// MI455X (gfx1250) — matrix operand order repaired and checked by reading, not compiled, not run on hardware
//
#include <hip/hip_runtime.h>
#include <math.h>

// ---------------------------------------------------------------------------
// CDNA5 (gfx1250) fused transformer block with pairwise causal expert.
// GEMMs: v_wmma_f32_16x16x32_bf16 (wave32).
// Tile staging: global_load_async_to_lds_b128 + s_wait_asynccnt (ASYNCcnt path).
// ---------------------------------------------------------------------------

typedef __bf16 bf16;
typedef bf16  v16bf __attribute__((ext_vector_type(16)));
typedef float v8f   __attribute__((ext_vector_type(8)));

static constexpr int Dm   = 512;   // model dim
static constexpr int Hh   = 8;     // heads
static constexpr int DHd  = 64;    // head dim
static constexpr int Bb2  = 2;     // batch
static constexpr int Ll   = 384;   // sequence
static constexpr int DFFc = 2048;  // ffn dim
static constexpr int ROWS = Bb2 * Ll;     // 768
static constexpr int NJC  = 8;            // j-chunks in expert
static constexpr int JCH  = 48;           // j per chunk (NJC*JCH == Ll)
static constexpr float EPSc = 1e-5f;

union Frag { uint4 u[2]; v16bf v; };

__device__ __forceinline__ float gelu_ex(float x) {
  return 0.5f * x * (1.0f + erff(x * 0.70710678118654752440f));
}

// Async 16B copy global -> LDS (per-lane), tracked by ASYNCcnt.
// LDS operand = low 32 bits of the flat shared pointer (ISA: LDS aperture
// keeps the DS byte offset in addr[31:0]).
__device__ __forceinline__ void async_cp16(const void* gsrc, void* ldst) {
  unsigned lds = (unsigned)(unsigned long long)(uintptr_t)ldst;
  unsigned long long ga = (unsigned long long)(uintptr_t)gsrc;
  asm volatile("global_load_async_to_lds_b128 %0, %1, off"
               :: "v"(lds), "v"(ga) : "memory");
}
__device__ __forceinline__ void async_wait0() {
  asm volatile("s_wait_asynccnt 0x0" ::: "memory");
}

// ---------------------------------------------------------------------------
// Generic bf16 WMMA GEMM.  A: MxK row-major.  B: NxK row-major (pre-transposed).
// Tile 64x64x32, 256 threads = 8 waves; wave w: m-subtile w&3, n-subtiles (w>>2)*2+{0,1}.
// ---------------------------------------------------------------------------
struct GemmP {
  const bf16* A;  long long as;  int lda;   // M x K row-major
  const bf16* Bm; long long bs;  int ldb;   // N x K row-major
  const float* bias;                        // length N or null
  const float* resid; long long rs; int ldr;
  float* Cf; long long cs; int ldc;         // fp32 out or null
  bf16*  Ch; long long chs; int ldch;       // bf16 out or null
  int omode;        // 0 plain, 1 qkv head-split, 2 attn-out merge, 3 v-transposed
  int act;          // 0 none, 1 gelu
  float scale;
  int M, N, K;
};

__global__ __launch_bounds__(256) void k_gemm(GemmP p) {
  __shared__ bf16 sA[64][40];   // [m][k], +8 pad
  __shared__ bf16 sB[64][40];   // [n][k], +8 pad

  const int t = threadIdx.x;
  const int w = t >> 5, l = t & 31;
  const int lhalf = l >> 4, ll = l & 15;
  const int msub = w & 3, ngrp = w >> 2;
  const int z  = blockIdx.z;
  const int m0 = blockIdx.y * 64, n0 = blockIdx.x * 64;

  const bf16* Ab = p.A  + (long long)z * p.as;
  const bf16* Bp = p.Bm + (long long)z * p.bs;

  const int ra = t >> 2, ka = (t & 3) * 8;   // A-tile: 64 rows x 32 cols
  v8f acc[2] = {};

  for (int k0 = 0; k0 < p.K; k0 += 32) {
    __syncthreads();
    async_cp16(Ab + (long long)(m0 + ra) * p.lda + k0 + ka, &sA[ra][ka]);
    async_cp16(Bp + (long long)(n0 + ra) * p.ldb + k0 + ka, &sB[ra][ka]);
    if (k0 + 32 < p.K)   // prefetch next streamed B chunk -> global_prefetch_b8
      __builtin_prefetch(Bp + (long long)(n0 + ra) * p.ldb + k0 + 32 + ka, 0, 1);
    async_wait0();
    __syncthreads();

    // A fragment: lane<16 -> K 0..7 & 16..23 ; lane>=16 -> +8 (ISA 16-bit A layout)
    Frag fa;
    fa.u[0] = *(const uint4*)&sA[msub * 16 + ll][lhalf * 8];
    fa.u[1] = *(const uint4*)&sA[msub * 16 + ll][16 + lhalf * 8];
#pragma unroll
    for (int j = 0; j < 2; ++j) {
      int n = (ngrp * 2 + j) * 16 + ll;
      Frag fb;          // lanes 0-15: K 0..7 & 16..23 ; lanes 16-31: +8
      fb.u[0] = *(const uint4*)&sB[n][lhalf * 8];
      fb.u[1] = *(const uint4*)&sB[n][16 + lhalf * 8];
      acc[j] = __builtin_amdgcn_wmma_f32_16x16x32_bf16(
          false, fa.v, false, fb.v, (short)0, acc[j], false, false);
    }
  }

  // epilogue: D element r of lane l = row (l>=16?8:0)+r, col l&15
#pragma unroll
  for (int j = 0; j < 2; ++j) {
    int colg = n0 + (ngrp * 2 + j) * 16 + ll;
    float bv = p.bias ? p.bias[colg] : 0.0f;
#pragma unroll
    for (int r = 0; r < 8; ++r) {
      int rowg = m0 + msub * 16 + lhalf * 8 + r;
      float v = p.scale * acc[j][r] + bv;
      if (p.act == 1) v = gelu_ex(v);
      if (p.omode == 0) {
        if (p.resid) v += p.resid[(long long)z * p.rs + (long long)rowg * p.ldr + colg];
        if (p.Cf) p.Cf[(long long)z * p.cs + (long long)rowg * p.ldc + colg] = v;
        if (p.Ch) p.Ch[(long long)z * p.chs + (long long)rowg * p.ldch + colg] = (bf16)v;
      } else if (p.omode == 1) {       // (b,i)x(h,d) -> [B,H,L,dh]
        int b = rowg / Ll, i = rowg % Ll, hh = colg >> 6, d = colg & 63;
        p.Ch[(((long long)(b * Hh + hh)) * Ll + i) * DHd + d] = (bf16)v;
      } else if (p.omode == 2) {       // batch (b,h) -> [B,L,D] at col h*64
        int b = z / Hh, hh = z % Hh;
        p.Ch[(long long)b * Ll * Dm + (long long)rowg * Dm + hh * DHd + colg] = (bf16)v;
      } else {                         // omode 3: (b,i)x(h,d) -> vT [B,H,dh,L]
        int b = rowg / Ll, i = rowg % Ll, hh = colg >> 6, d = colg & 63;
        p.Ch[(((long long)(b * Hh + hh)) * DHd + d) * Ll + i] = (bf16)v;
      }
    }
  }
}

// ---------------------------------------------------------------------------
// Fused pairwise causal expert.
// Block = (jc, iblk, b): 16 query rows i0..i0+15, sweeps j in [jc*48, ...).
// Per j: sA = gelu(A_i + B_j + cb1) (16x512 bf16), GEMM vs cw2T (NxK bf16),
// +cb2, row LayerNorm, tril mask, accumulate into LDS; partial -> featp[jc].
// ---------------------------------------------------------------------------
__global__ __launch_bounds__(256) void k_expert(
    const float* __restrict__ Abuf, const float* __restrict__ Bmbuf,
    const float* __restrict__ cb1,  const bf16*  __restrict__ cw2T,
    const float* __restrict__ cb2,  const float* __restrict__ clng,
    const float* __restrict__ clnb, float* __restrict__ featp) {
  __shared__ bf16 sA[16][520];                       // act tile, full K, +8 pad
  __shared__ __align__(16) char smem[512 * 40 * 2];  // sB [512][40] bf16, aliased by sU [16][512] f32
  __shared__ float facc[16][Dm];
  __shared__ float sSum[16][16], sSsq[16][16], sMu[16], sRs[16];
  bf16  (*sB)[40] = reinterpret_cast<bf16(*)[40]>(smem);
  float (*sU)[Dm] = reinterpret_cast<float(*)[Dm]>(smem);

  const int t = threadIdx.x;
  const int w = t >> 5, l = t & 31, lhalf = l >> 4, ll = l & 15;
  const int jc = blockIdx.x, iblk = blockIdx.y, b = blockIdx.z;
  const int i0 = iblk * 16;
  const int jmax = i0 + 16;                 // valid j satisfies j <= i <= i0+15
  const int j0 = jc * JCH;
  const int j1 = (j0 + JCH < jmax) ? (j0 + JCH) : jmax;
  if (j0 >= jmax) return;                   // featp pre-zeroed; nothing to add

  for (int it = 0; it < 32; ++it) {         // zero LDS accumulator
    int idx = t + it * 256;
    facc[idx >> 9][idx & 511] = 0.0f;
  }

  const float* Arow0 = Abuf + ((long long)b * Ll + i0) * Dm;

  for (int j = j0; j < j1; ++j) {
    __syncthreads();
    const float* Brow = Bmbuf + ((long long)b * Ll + j) * Dm;
    for (int it = 0; it < 32; ++it) {       // stage activations (16x512)
      int idx = t + it * 256;
      int r = idx >> 9, k = idx & 511;
      float v = Arow0[(long long)r * Dm + k] + Brow[k] + cb1[k];
      sA[r][k] = (bf16)gelu_ex(v);
    }

    v8f acc[4] = {};
    for (int ks = 0; ks < 16; ++ks) {
      __syncthreads();
      int k0 = ks * 32;
#pragma unroll
      for (int it = 0; it < 8; ++it) {      // cw2T slice 512 x 32 -> sB[n][k], async DMA
        int c = t + it * 256;               // 2048 x 16B chunks
        int n = c >> 2, kq = (c & 3) * 8;
        async_cp16(cw2T + (long long)n * Dm + k0 + kq, &sB[n][kq]);
      }
      if (ks + 1 < 16)
        __builtin_prefetch(cw2T + (long long)(t << 1) * Dm + k0 + 32, 0, 1);
      async_wait0();
      __syncthreads();

      Frag fa;
      fa.u[0] = *(const uint4*)&sA[ll][k0 + lhalf * 8];
      fa.u[1] = *(const uint4*)&sA[ll][k0 + 16 + lhalf * 8];
#pragma unroll
      for (int s = 0; s < 4; ++s) {
        int n = w * 64 + s * 16 + ll;
        Frag fb;
        fb.u[0] = *(const uint4*)&sB[n][lhalf * 8];
        fb.u[1] = *(const uint4*)&sB[n][16 + lhalf * 8];
        acc[s] = __builtin_amdgcn_wmma_f32_16x16x32_bf16(
            false, fa.v, false, fb.v, (short)0, acc[s], false, false);
      }
    }
    __syncthreads();                        // all sB reads done; reuse as sU
#pragma unroll
    for (int s = 0; s < 4; ++s) {
      int col = w * 64 + s * 16 + ll;
      float cb = cb2[col];
#pragma unroll
      for (int r = 0; r < 8; ++r) sU[lhalf * 8 + r][col] = acc[s][r] + cb;
    }
    __syncthreads();
    { // LN stats: 16 threads per row, 32 elems each
      int r = t >> 4, q = t & 15;
      float s = 0.0f, ss = 0.0f;
      const float* u = &sU[r][q * 32];
#pragma unroll
      for (int k = 0; k < 32; ++k) { float v = u[k]; s += v; ss += v * v; }
      sSum[r][q] = s; sSsq[r][q] = ss;
    }
    __syncthreads();
    if (t < 16) {
      float s = 0.0f, ss = 0.0f;
      for (int q = 0; q < 16; ++q) { s += sSum[t][q]; ss += sSsq[t][q]; }
      float mu = s * (1.0f / 512.0f);
      float var = ss * (1.0f / 512.0f) - mu * mu;
      sMu[t] = mu; sRs[t] = rsqrtf(var + EPSc);
    }
    __syncthreads();
    for (int it = 0; it < 32; ++it) {       // masked accumulate (deterministic order)
      int idx = t + it * 256;
      int r = idx >> 9, k = idx & 511;
      if (j <= i0 + r) {
        float v = (sU[r][k] - sMu[r]) * sRs[r] * clng[k] + clnb[k];
        facc[r][k] += v;
      }
    }
  }
  __syncthreads();
  for (int it = 0; it < 32; ++it) {         // store this chunk's partial
    int idx = t + it * 256;
    int r = idx >> 9, k = idx & 511;
    featp[((long long)jc * ROWS + (long long)b * Ll + i0 + r) * Dm + k] = facc[r][k];
  }
}

// ---------------------------------------------------------------------------
// Small kernels
// ---------------------------------------------------------------------------
__global__ __launch_bounds__(256) void k_add_te(const float* __restrict__ x,
                                                const float* __restrict__ te,
                                                float* __restrict__ out) {
  int idx = blockIdx.x * 256 + threadIdx.x;   // ROWS*Dm exact
  int row = idx / Dm, k = idx % Dm, i = row % Ll;
  out[idx] = x[idx] + te[(long long)i * Dm + k];
}

// fp32 KxN -> bf16 NxK (transpose + convert); one-time per launch.
__global__ __launch_bounds__(256) void k_cvt_t(const float* __restrict__ in,
                                               bf16* __restrict__ out, int K, int N) {
  int idx = blockIdx.x * 256 + threadIdx.x;
  if (idx < N * K) {
    int n = idx / K, k = idx % K;
    out[idx] = (bf16)in[(long long)k * N + n];
  }
}

__global__ __launch_bounds__(256) void k_zero(float* __restrict__ p, int n) {
  int i = blockIdx.x * 256 + threadIdx.x;
  if (i < n) p[i] = 0.0f;
}

__global__ __launch_bounds__(256) void k_resid(const float* __restrict__ x1,
                                               const float* __restrict__ featp,
                                               float* __restrict__ x2) {
  long long idx = (long long)blockIdx.x * 256 + threadIdx.x;  // ROWS*Dm exact
  float s = 0.0f;
  for (int jc = 0; jc < NJC; ++jc) s += featp[(long long)jc * ROWS * Dm + idx];
  x2[idx] = x1[idx] + s * (1.0f / (float)Ll);
}

__global__ __launch_bounds__(256) void k_ln(const float* __restrict__ x,
                                            const float* __restrict__ g,
                                            const float* __restrict__ bet,
                                            bf16* __restrict__ out, int rows) {
  __shared__ float sS[256], sQ[256];
  int r = blockIdx.x;
  if (r >= rows) return;
  const float* xr = x + (long long)r * Dm;
  int t = threadIdx.x;
  float a = xr[t], c = xr[t + 256];
  sS[t] = a + c; sQ[t] = a * a + c * c;
  __syncthreads();
  for (int o = 128; o > 0; o >>= 1) {
    if (t < o) { sS[t] += sS[t + o]; sQ[t] += sQ[t + o]; }
    __syncthreads();
  }
  float mu = sS[0] * (1.0f / 512.0f);
  float var = sQ[0] * (1.0f / 512.0f) - mu * mu;
  float rs = rsqrtf(var + EPSc);
  out[(long long)r * Dm + t]       = (bf16)((a - mu) * rs * g[t] + bet[t]);
  out[(long long)r * Dm + t + 256] = (bf16)((c - mu) * rs * g[t + 256] + bet[t + 256]);
}

__global__ __launch_bounds__(256) void k_softmax(const float* __restrict__ scores,
                                                 bf16* __restrict__ attn) {
  __shared__ float red[256];
  int row = blockIdx.x;                 // B*H*L rows
  int i = row % Ll;
  const float* s = scores + (long long)row * Ll;
  bf16* a = attn + (long long)row * Ll;
  int t = threadIdx.x;
  float m = -1e30f;
  for (int j = t; j < Ll; j += 256) if (j <= i) m = fmaxf(m, s[j]);
  red[t] = m; __syncthreads();
  for (int o = 128; o > 0; o >>= 1) { if (t < o) red[t] = fmaxf(red[t], red[t + o]); __syncthreads(); }
  m = red[0]; __syncthreads();
  float sum = 0.0f;
  for (int j = t; j < Ll; j += 256) if (j <= i) sum += expf(s[j] - m);
  red[t] = sum; __syncthreads();
  for (int o = 128; o > 0; o >>= 1) { if (t < o) red[t] += red[t + o]; __syncthreads(); }
  float inv = 1.0f / red[0];
  for (int j = t; j < Ll; j += 256)
    a[j] = (bf16)((j <= i) ? expf(s[j] - m) * inv : 0.0f);
}

// ---------------------------------------------------------------------------
// Host orchestration
// ---------------------------------------------------------------------------
extern "C" void kernel_launch(void* const* d_in, const int* in_sizes, int n_in,
                              void* d_out, int out_size, void* d_ws, size_t ws_size,
                              hipStream_t stream) {
  const float* x    = (const float*)d_in[0];
  const float* te   = (const float*)d_in[1];
  const float* n1g  = (const float*)d_in[2];
  const float* n1b  = (const float*)d_in[3];
  const float* wq   = (const float*)d_in[4];
  const float* bq   = (const float*)d_in[5];
  const float* wk   = (const float*)d_in[6];
  const float* bk   = (const float*)d_in[7];
  const float* wv   = (const float*)d_in[8];
  const float* bv   = (const float*)d_in[9];
  const float* wo   = (const float*)d_in[10];
  const float* bo   = (const float*)d_in[11];
  const float* cw1  = (const float*)d_in[12];
  const float* cb1  = (const float*)d_in[13];
  const float* cw2  = (const float*)d_in[14];
  const float* cb2  = (const float*)d_in[15];
  const float* clng = (const float*)d_in[16];
  const float* clnb = (const float*)d_in[17];
  const float* n2g  = (const float*)d_in[18];
  const float* n2b  = (const float*)d_in[19];
  const float* ew1  = (const float*)d_in[20];
  const float* eb1  = (const float*)d_in[21];
  const float* ew2  = (const float*)d_in[22];
  const float* eb2  = (const float*)d_in[23];
  float* out = (float*)d_out;

  char* ws = (char*)d_ws;
  size_t off = 0;
  auto carve = [&](size_t bytes) -> void* {
    void* p = ws + off; off = (off + bytes + 255) & ~(size_t)255; return p;
  };
  float* xa    = (float*)carve((size_t)ROWS * Dm * 4);
  bf16*  hbf   = (bf16*) carve((size_t)ROWS * Dm * 2);
  bf16*  wqT   = (bf16*) carve((size_t)Dm * Dm * 2);
  bf16*  wkT   = (bf16*) carve((size_t)Dm * Dm * 2);
  bf16*  wvT   = (bf16*) carve((size_t)Dm * Dm * 2);
  bf16*  woT   = (bf16*) carve((size_t)Dm * Dm * 2);
  bf16*  w1aT  = (bf16*) carve((size_t)Dm * Dm * 2);
  bf16*  w1bT  = (bf16*) carve((size_t)Dm * Dm * 2);
  bf16*  cw2T  = (bf16*) carve((size_t)Dm * Dm * 2);
  bf16*  ew1T  = (bf16*) carve((size_t)Dm * DFFc * 2);
  bf16*  ew2T  = (bf16*) carve((size_t)DFFc * Dm * 2);
  bf16*  qb    = (bf16*) carve((size_t)Bb2 * Hh * Ll * DHd * 2);
  bf16*  kb    = (bf16*) carve((size_t)Bb2 * Hh * Ll * DHd * 2);
  bf16*  vbT   = (bf16*) carve((size_t)Bb2 * Hh * DHd * Ll * 2);
  float* scr   = (float*)carve((size_t)Bb2 * Hh * Ll * Ll * 4);
  bf16*  attn  = (bf16*) carve((size_t)Bb2 * Hh * Ll * Ll * 2);
  bf16*  obuf  = (bf16*) carve((size_t)ROWS * Dm * 2);
  float* x1    = (float*)carve((size_t)ROWS * Dm * 4);
  bf16*  x1h   = (bf16*) carve((size_t)ROWS * Dm * 2);
  float* Abuf  = (float*)carve((size_t)ROWS * Dm * 4);
  float* Bmbuf = (float*)carve((size_t)ROWS * Dm * 4);
  float* featp = (float*)carve((size_t)NJC * ROWS * Dm * 4);
  float* x2    = (float*)carve((size_t)ROWS * Dm * 4);
  bf16*  h2    = (bf16*) carve((size_t)ROWS * Dm * 2);
  bf16*  g1    = (bf16*) carve((size_t)ROWS * DFFc * 2);
  (void)ws_size; (void)in_sizes; (void)n_in; (void)out_size;

  auto cvtT = [&](const float* in, bf16* o, int K, int N) {
    k_cvt_t<<<(N * K + 255) / 256, 256, 0, stream>>>(in, o, K, N);
  };
  auto gemm = [&](const bf16* A, long long as, int lda,
                  const bf16* Bp, long long bs, int ldb,
                  const float* bias, const float* resid, long long rs, int ldr,
                  float* Cf, long long cs, int ldc,
                  bf16* Ch, long long chs, int ldch,
                  int omode, int act, float scale, int M, int N, int K, int nb) {
    GemmP p;
    p.A = A; p.as = as; p.lda = lda;
    p.Bm = Bp; p.bs = bs; p.ldb = ldb;
    p.bias = bias; p.resid = resid; p.rs = rs; p.ldr = ldr;
    p.Cf = Cf; p.cs = cs; p.ldc = ldc;
    p.Ch = Ch; p.chs = chs; p.ldch = ldch;
    p.omode = omode; p.act = act; p.scale = scale;
    p.M = M; p.N = N; p.K = K;
    dim3 g(N / 64, M / 64, nb);
    k_gemm<<<g, 256, 0, stream>>>(p);
  };

  // 0) weights -> bf16, transposed to NxK (one-time)
  cvtT(wq,  wqT,  Dm, Dm);
  cvtT(wk,  wkT,  Dm, Dm);
  cvtT(wv,  wvT,  Dm, Dm);
  cvtT(wo,  woT,  Dm, Dm);
  cvtT(cw1,                   w1aT, Dm, Dm);
  cvtT(cw1 + (size_t)Dm * Dm, w1bT, Dm, Dm);
  cvtT(cw2, cw2T, Dm, Dm);
  cvtT(ew1, ew1T, Dm, DFFc);
  cvtT(ew2, ew2T, DFFc, Dm);

  // 1) x + temporal_enc
  k_add_te<<<ROWS * Dm / 256, 256, 0, stream>>>(x, te, xa);
  // 2) LN1 -> bf16
  k_ln<<<ROWS, 256, 0, stream>>>(xa, n1g, n1b, hbf, ROWS);

  const long long HS = (long long)Ll * DHd;  // per-(b,h) q/k/v stride
  // 3) QKV projections (q,k head-split; v head-transposed)
  gemm(hbf, 0, Dm, wqT, 0, Dm, bq, nullptr, 0, 0,
       nullptr, 0, 0, qb, 0, 0, 1, 0, 1.0f, ROWS, Dm, Dm, 1);
  gemm(hbf, 0, Dm, wkT, 0, Dm, bk, nullptr, 0, 0,
       nullptr, 0, 0, kb, 0, 0, 1, 0, 1.0f, ROWS, Dm, Dm, 1);
  gemm(hbf, 0, Dm, wvT, 0, Dm, bv, nullptr, 0, 0,
       nullptr, 0, 0, vbT, 0, 0, 3, 0, 1.0f, ROWS, Dm, Dm, 1);
  // 4) scores = q k^T / 8   (batched over B*H; k already NxK = L x dh)
  gemm(qb, HS, DHd, kb, HS, DHd, nullptr, nullptr, 0, 0,
       scr, (long long)Ll * Ll, Ll, nullptr, 0, 0, 0, 0, 0.125f,
       Ll, Ll, DHd, Bb2 * Hh);
  // 5) causal softmax -> bf16
  k_softmax<<<Bb2 * Hh * Ll, 256, 0, stream>>>(scr, attn);
  // 6) o = attn @ v  (B-operand = vT, NxK = dh x L), merged into [B,L,D]
  gemm(attn, (long long)Ll * Ll, Ll, vbT, HS, Ll, nullptr, nullptr, 0, 0,
       nullptr, 0, 0, obuf, 0, 0, 2, 0, 1.0f, Ll, DHd, Ll, Bb2 * Hh);
  // 7) x1 = xa + o @ wo + bo
  gemm(obuf, 0, Dm, woT, 0, Dm, bo, xa, 0, Dm,
       x1, 0, Dm, x1h, 0, Dm, 0, 0, 1.0f, ROWS, Dm, Dm, 1);
  // 8) A = x1 @ w1a ; Bm = x1 @ w1b (bias cb1 folded into expert)
  gemm(x1h, 0, Dm, w1aT, 0, Dm, nullptr, nullptr, 0, 0,
       Abuf, 0, Dm, nullptr, 0, 0, 0, 0, 1.0f, ROWS, Dm, Dm, 1);
  gemm(x1h, 0, Dm, w1bT, 0, Dm, nullptr, nullptr, 0, 0,
       Bmbuf, 0, Dm, nullptr, 0, 0, 0, 0, 1.0f, ROWS, Dm, Dm, 1);
  // 9) fused causal expert (deterministic j-chunk partials)
  k_zero<<<(NJC * ROWS * Dm + 255) / 256, 256, 0, stream>>>(featp, NJC * ROWS * Dm);
  {
    dim3 g(NJC, Ll / 16, Bb2);
    k_expert<<<g, 256, 0, stream>>>(Abuf, Bmbuf, cb1, cw2T, cb2, clng, clnb, featp);
  }
  // 10) x2 = x1 + mean_j(rel)
  k_resid<<<ROWS * Dm / 256, 256, 0, stream>>>(x1, featp, x2);
  // 11) FFN
  k_ln<<<ROWS, 256, 0, stream>>>(x2, n2g, n2b, h2, ROWS);
  gemm(h2, 0, Dm, ew1T, 0, Dm, eb1, nullptr, 0, 0,
       nullptr, 0, 0, g1, 0, DFFc, 0, 1, 1.0f, ROWS, DFFc, Dm, 1);
  gemm(g1, 0, DFFc, ew2T, 0, DFFc, eb2, x2, 0, Dm,
       out, 0, Dm, nullptr, 0, 0, 0, 0, 1.0f, ROWS, Dm, DFFc, 1);
}
